// ConvexMultiHeadAttention_88527865905583
// MI455X (gfx1250) — matrix operand order repaired and checked by reading, not compiled, not run on hardware
//
#include <hip/hip_runtime.h>

// ---------------------------------------------------------------------------
// ConvexMultiHeadAttention for MI455X (gfx1250, wave32, WMMA bf16 16x16x32)
//
//   B=2, S=2048, D=1024, H=16, hd=64
//   out = (sum_t n(z_t) * V_t) / (sum_t n(z_t) + eps),
//   n(z) = relu(exp(clip(z,-15,15) - 1) + 0.1*z)      [no max-rescale needed]
//
// Kernel 1: X@W+b in bf16 WMMA, fused d_q/d_k/d_v scaling. Writes bf16
//           Q,K row-major per head and V TRANSPOSED per head ([bh][d][s]) so
//           the attention kernel can stage V by contiguous async copies.
// Kernel 2: single-pass fused attention; K/V tiles staged into LDS with
//           double-buffered GLOBAL_LOAD_ASYNC_TO_LDS_B128 (ASYNCcnt), the
//           denominator computed with a ones-column WMMA.
// Workspace: 3 * 2*16*2048*64 bf16 = 24 MB (resident in 192 MB L2).
// ---------------------------------------------------------------------------

typedef __attribute__((ext_vector_type(16))) __bf16        v16bf;
typedef __attribute__((ext_vector_type(8)))  float         v8f;
typedef __attribute__((ext_vector_type(4)))  unsigned int  v4u;
typedef __attribute__((ext_vector_type(4)))  float         v4f;

union Frag {
    v16bf          v;      // WMMA operand (16 bf16 per lane)
    unsigned short s[16];
    v4u            q[2];   // two 16B chunks
};

__device__ __forceinline__ unsigned short f2bf(float f) {
    unsigned int u = __builtin_bit_cast(unsigned int, f);
    u += 0x7FFFu + ((u >> 16) & 1u);               // round-to-nearest-even
    return (unsigned short)(u >> 16);
}
__device__ __forceinline__ unsigned int pk2(float a, float b) {
    return (unsigned int)f2bf(a) | ((unsigned int)f2bf(b) << 16);
}
__device__ __forceinline__ v8f wmma_bf16(const Frag& a, const Frag& b, v8f c) {
    return __builtin_amdgcn_wmma_f32_16x16x32_bf16(
        false, a.v, false, b.v, (short)0, c, false, false);
}

// --- gfx1250 async global->LDS copy (16B per lane), tracked by ASYNCcnt ----
__device__ __forceinline__ void async_copy16(unsigned lds_off, const void* gaddr) {
    asm volatile("global_load_async_to_lds_b128 %0, %1, off"
                 :: "v"(lds_off), "v"(gaddr)
                 : "memory");
}
__device__ __forceinline__ void wait_async0() {
    asm volatile("s_wait_asynccnt 0x0" ::: "memory");
}
__device__ __forceinline__ unsigned lds_off32(const void* p) {
    // flat address of LDS: addr[31:0] is the LDS byte address (ISA §10.2)
    return (unsigned)(unsigned long long)p;
}

#define D_MODEL 1024
#define SEQ     2048
#define HEADS   16
#define HD      64

// ===========================================================================
// Kernel 1: projection GEMM  X[4096x1024] @ W[1024x1024] + b, then *dq/dk/dv.
// Q,K: [bh][s][d] row-major.  V: [bh][d][s] (transposed).  All bf16.
// Block: 256 thr (8 waves). Tile: 64 rows x 64 cols (one full head of cols).
// ===========================================================================
__global__ void __launch_bounds__(256)
cvx_proj_kernel(const float* __restrict__ x,  const float* __restrict__ W,
                const float* __restrict__ bias,
                const float* __restrict__ dq, const float* __restrict__ dk,
                const float* __restrict__ dv,
                unsigned short* __restrict__ Qb, unsigned short* __restrict__ Kb,
                unsigned short* __restrict__ Vb)
{
    __shared__ __attribute__((aligned(16))) unsigned short As[64 * 32]; // [m][k]
    __shared__ __attribute__((aligned(16))) unsigned short Bs[64 * 32]; // [n][k] = W^T

    const int n0   = blockIdx.x * 64;          // column base (head h = n0/64)
    const int m0   = blockIdx.y * 64;          // row base in [0,4096)
    const int tid  = threadIdx.x;
    const int lane = tid & 31;
    const int wave = tid >> 5;
    const int m16  = (wave >> 1) * 16;         // wave row strip: 0/16/32/48
    const int n16  = (wave & 1) * 32;          // wave col strip: 0/32 (2 tiles)

    v8f acc0 = {}, acc1 = {};

    for (int k0 = 0; k0 < D_MODEL; k0 += 32) {
        __syncthreads();
        {   // stage x tile [64 x 32] f32 -> bf16 LDS (row major)
            const int r = tid >> 2, c = (tid & 3) * 8;
            const float* src = x + (size_t)(m0 + r) * D_MODEL + k0 + c;
            v4f f0 = *(const v4f*)src;
            v4f f1 = *(const v4f*)(src + 4);
            v4u pk = { pk2(f0[0], f0[1]), pk2(f0[2], f0[3]),
                       pk2(f1[0], f1[1]), pk2(f1[2], f1[3]) };
            *(v4u*)&As[r * 32 + c] = pk;
        }
        {   // stage W tile [32 x 64] transposed -> Bs[n][k] bf16
            const int kr = tid >> 3, nc = (tid & 7) * 8;
            const float* src = W + (size_t)(k0 + kr) * D_MODEL + n0 + nc;
            v4f f0 = *(const v4f*)src;
            v4f f1 = *(const v4f*)(src + 4);
            Bs[(nc + 0) * 32 + kr] = f2bf(f0[0]);
            Bs[(nc + 1) * 32 + kr] = f2bf(f0[1]);
            Bs[(nc + 2) * 32 + kr] = f2bf(f0[2]);
            Bs[(nc + 3) * 32 + kr] = f2bf(f0[3]);
            Bs[(nc + 4) * 32 + kr] = f2bf(f1[0]);
            Bs[(nc + 5) * 32 + kr] = f2bf(f1[1]);
            Bs[(nc + 6) * 32 + kr] = f2bf(f1[2]);
            Bs[(nc + 7) * 32 + kr] = f2bf(f1[3]);
        }
        __syncthreads();

        // A fragment: lane L -> row m16+(L&15); K = (L/16)*16..+15 contiguous
        Frag a;
        {
            const int ar = m16 + (lane & 15);
            const int ao = (lane >> 4) * 16;
            a.q[0] = *(const v4u*)&As[ar * 32 + ao];
            a.q[1] = *(const v4u*)&As[ar * 32 + ao + 8];
        }
        // B fragments: lane L -> col n16+t*16+(L&15); K = (L/16)*16..+15 contiguous
        Frag b0, b1;
        {
            const int bo = (lane >> 4) * 16;
            const int c0 = n16 + (lane & 15);
            b0.q[0] = *(const v4u*)&Bs[c0 * 32 + bo];
            b0.q[1] = *(const v4u*)&Bs[c0 * 32 + bo + 8];
            const int c1 = c0 + 16;
            b1.q[0] = *(const v4u*)&Bs[c1 * 32 + bo];
            b1.q[1] = *(const v4u*)&Bs[c1 * 32 + bo + 8];
        }
        acc0 = wmma_bf16(a, b0, acc0);
        acc1 = wmma_bf16(a, b1, acc1);
    }

    // Epilogue: +bias, *dq/dk/dv, bf16; Q/K row-major, V transposed per head
    const int colL   = lane & 15;
    const int rowOff = (lane >> 4) * 8;
#pragma unroll
    for (int t = 0; t < 2; ++t) {
        v8f acc = t ? acc1 : acc0;
        const int   ncol = n0 + n16 + t * 16 + colL;
        const float bb = bias[ncol], fq = dq[ncol], fk = dk[ncol], fv = dv[ncol];
        const int   h = ncol >> 6, dloc = ncol & 63;
#pragma unroll
        for (int j = 0; j < 8; ++j) {
            const int m = m0 + m16 + rowOff + j;
            const int bidx = m >> 11;          // batch
            const int s    = m & 2047;         // seq pos
            const float val = acc[j] + bb;
            const int bh = bidx * HEADS + h;
            const size_t qkBase = ((size_t)bh * SEQ + s) * HD + dloc;
            Qb[qkBase] = f2bf(val * fq);
            Kb[qkBase] = f2bf(val * fk);
            Vb[((size_t)bh * HD + dloc) * SEQ + s] = f2bf(val * fv);  // V^T
        }
    }
}

// ===========================================================================
// Kernel 2: fused convex attention. Block = 256 thr (8 waves) owns 128 query
// rows of one (b,h); each wave owns 16 rows. kv loop in 32-col tiles staged
// with double-buffered async global->LDS copies (one b128 per lane per tile).
// ===========================================================================
__global__ void __launch_bounds__(256)
cvx_attn_kernel(const unsigned short* __restrict__ Qb,
                const unsigned short* __restrict__ Kb,
                const unsigned short* __restrict__ Vb,
                float* __restrict__ out)
{
    __shared__ __attribute__((aligned(16))) unsigned short Ks[2][32 * 64]; // [t][d]
    __shared__ __attribute__((aligned(16))) unsigned short Vt[2][64 * 32]; // [d][t]
    __shared__ __attribute__((aligned(16))) unsigned short Ps[8 * 16 * 32]; // P bounce

    const int q0   = blockIdx.x * 128;
    const int h    = blockIdx.y;
    const int bz   = blockIdx.z;
    const int tid  = threadIdx.x;
    const int lane = tid & 31;
    const int wave = tid >> 5;
    const int bh   = bz * HEADS + h;
    const unsigned short* Kg = Kb + (size_t)bh * SEQ * HD;   // [s][d]
    const unsigned short* Vg = Vb + (size_t)bh * HD * SEQ;   // [d][s]
    const int qrow = q0 + wave * 16;

    // Per-thread async staging slots (16B each; 256 threads cover 4KB tiles)
    const int kR = tid >> 3, kC = (tid & 7) * 8;   // K tile: [32 t][64 d]
    const int vD = tid >> 2, vC = (tid & 3) * 8;   // V tile: [64 d][32 t]

    // Q fragments (16 rows x 64 k -> two 16x32 A frags), loaded once
    Frag aQ0, aQ1;
    {
        const int r = qrow + (lane & 15);
        const int o = (lane >> 4) * 16;
        const unsigned short* qp = Qb + (size_t)bh * SEQ * HD + (size_t)r * HD;
        aQ0.q[0] = *(const v4u*)(qp + o);
        aQ0.q[1] = *(const v4u*)(qp + o + 8);
        aQ1.q[0] = *(const v4u*)(qp + 32 + o);
        aQ1.q[1] = *(const v4u*)(qp + 32 + o + 8);
    }
    // B fragment of the "ones column" tile (col 0 = 1.0, else 0)
    Frag bOne;
#pragma unroll
    for (int i = 0; i < 16; ++i)
        bOne.s[i] = ((lane & 15) == 0) ? (unsigned short)0x3F80 : (unsigned short)0;

    v8f accO[4] = {v8f{}, v8f{}, v8f{}, v8f{}};
    v8f accD = {};
    unsigned short* Pw = &Ps[wave * 16 * 32];

    // Prefetch tile 0 into buffer 0
    async_copy16(lds_off32(&Ks[0][kR * 64 + kC]), Kg + (size_t)kR * HD + kC);
    async_copy16(lds_off32(&Vt[0][vD * 32 + vC]), Vg + (size_t)vD * SEQ + vC);

    for (int t0 = 0; t0 < SEQ; t0 += 32) {
        const int bi = (t0 >> 5) & 1;
        wait_async0();                    // this wave's prefetch into buf bi done
        __syncthreads();                  // -> all waves' prefetches visible; all
                                          //    waves done reading buf bi^1
        if (t0 + 32 < SEQ) {              // prefetch next tile into other buffer
            const int t1 = t0 + 32;
            async_copy16(lds_off32(&Ks[bi ^ 1][kR * 64 + kC]),
                         Kg + (size_t)(t1 + kR) * HD + kC);
            async_copy16(lds_off32(&Vt[bi ^ 1][vD * 32 + vC]),
                         Vg + (size_t)vD * SEQ + t1 + vC);
        }
        const unsigned short* Kt = Ks[bi];
        const unsigned short* Vv = Vt[bi];

        // --- scores S = Q K^T for two 16-col t-chunks (K-dim 64 = 2 WMMAs) ---
        v8f st[2];
#pragma unroll
        for (int tc = 0; tc < 2; ++tc) {
            Frag bK0, bK1;
            const int col = tc * 16 + (lane & 15);   // local t index (= B col)
            const int o   = (lane >> 4) * 16;        // d offset for this lane half
            bK0.q[0] = *(const v4u*)&Kt[col * 64 + o];
            bK0.q[1] = *(const v4u*)&Kt[col * 64 + o + 8];
            bK1.q[0] = *(const v4u*)&Kt[col * 64 + 32 + o];
            bK1.q[1] = *(const v4u*)&Kt[col * 64 + 32 + o + 8];
            v8f s = {};
            s = wmma_bf16(aQ0, bK0, s);
            s = wmma_bf16(aQ1, bK1, s);
            st[tc] = s;
        }

        // --- convex nonlinearity, P -> wave-private LDS (DS ops in-order) ---
#pragma unroll
        for (int tc = 0; tc < 2; ++tc) {
            const int mOff = (lane >> 4) * 8;
            const int n    = tc * 16 + (lane & 15);
#pragma unroll
            for (int j = 0; j < 8; ++j) {
                float z = st[tc][j];
                z = fminf(fmaxf(z, -15.0f), 15.0f);
                float p = __expf(z - 1.0f) + 0.1f * z;   // exp(z-R) + LAM*z
                p = fmaxf(p, 0.0f);                       // relu
                Pw[(mOff + j) * 32 + n] = f2bf(p);
            }
        }
        // Re-read as A fragment (16 rows x 32 k of attention weights)
        Frag aP;
        {
            const int r = lane & 15;
            const int o = (lane >> 4) * 16;
            aP.q[0] = *(const v4u*)&Pw[r * 32 + o];
            aP.q[1] = *(const v4u*)&Pw[r * 32 + o + 8];
        }

        // --- accumulate P@V (4 n-tiles) and denominator (ones tile) ---
#pragma unroll
        for (int nt = 0; nt < 4; ++nt) {
            Frag bV;
            const int col = nt * 16 + (lane & 15);   // d index (= B col)
            const int o   = (lane >> 4) * 16;        // t offset
            bV.q[0] = *(const v4u*)&Vv[col * 32 + o];
            bV.q[1] = *(const v4u*)&Vv[col * 32 + o + 8];
            accO[nt] = wmma_bf16(aP, bV, accO[nt]);
        }
        accD = wmma_bf16(aP, bOne, accD);
    }

    // Epilogue: out = accO / (den + eps). den for row j (+8) sits in lane 0 (16).
    const int srcLane = (lane < 16) ? 0 : 16;
#pragma unroll
    for (int j = 0; j < 8; ++j) {
        const float den = __shfl(accD[j], srcLane, 32);
        const float inv = 1.0f / (den + 1e-9f);
        const int m = qrow + (lane >> 4) * 8 + j;     // seq row
        float* op = out + ((size_t)(bz * SEQ + m)) * D_MODEL + h * HD;
#pragma unroll
        for (int nt = 0; nt < 4; ++nt)
            op[nt * 16 + (lane & 15)] = accO[nt][j] * inv;
    }
}

// ===========================================================================
extern "C" void kernel_launch(void* const* d_in, const int* in_sizes, int n_in,
                              void* d_out, int out_size, void* d_ws, size_t ws_size,
                              hipStream_t stream)
{
    const float* x    = (const float*)d_in[0];
    const float* W    = (const float*)d_in[1];
    const float* bias = (const float*)d_in[2];
    const float* dq   = (const float*)d_in[3];
    const float* dk   = (const float*)d_in[4];
    const float* dv   = (const float*)d_in[5];
    float* out = (float*)d_out;

    const size_t per = (size_t)2 * HEADS * SEQ * HD;     // 4M bf16 elems = 8 MB
    unsigned short* Qb = (unsigned short*)d_ws;
    unsigned short* Kb = Qb + per;
    unsigned short* Vb = Kb + per;

    dim3 gp(D_MODEL / 64, (2 * SEQ) / 64);               // 16 x 64 blocks
    cvx_proj_kernel<<<gp, dim3(256), 0, stream>>>(x, W, bias, dq, dk, dv,
                                                  Qb, Kb, Vb);

    dim3 ga(SEQ / 128, HEADS, 2);                        // 16 x 16 x 2 blocks
    cvx_attn_kernel<<<ga, dim3(256), 0, stream>>>(Qb, Kb, Vb, out);
}
